// FlashSVDLlamaBlock_26980984553790
// MI455X (gfx1250) — compile-verified, matrix operand order checked by reading, not run on hardware
//
#include <hip/hip_runtime.h>
#include <hip/hip_bf16.h>
#include <math.h>

// ---------------------------------------------------------------------------
// FlashSVD Llama block for gfx1250 (MI455X).  All matmuls run through
// v_wmma_f32_16x16x32_f16 (wave32 WMMA); V tiles for attention are staged
// into LDS with GLOBAL_LOAD_ASYNC_TO_LDS_B128 (ASYNCcnt).
// Shapes: B=2 T=1024 D=2048 H=32 HK=8 DH=64 RQ=RKV=48(pad 64) RO=RFF=1024
//         INTER=8192
// ---------------------------------------------------------------------------

typedef __attribute__((ext_vector_type(16))) _Float16 v16h;
typedef __attribute__((ext_vector_type(8)))  _Float16 v8h;
typedef __attribute__((ext_vector_type(8)))  float    v8f;

__device__ __forceinline__ v8f zero8f() {
    v8f v;
#pragma unroll
    for (int i = 0; i < 8; ++i) v[i] = 0.0f;
    return v;
}

// Load a 16x32 f16 fragment (A-layout; also valid B-layout for N x K
// row-major matrices): lane L holds row (L&15); VGPR k-offset base is
// ((L>>4)*8) with the documented {0..7, 16..23} split.
__device__ __forceinline__ v16h load_frag_rowmajor(const _Float16* __restrict__ base,
                                                   int ld, int row0, int k0, int lane) {
    int r  = row0 + (lane & 15);
    int kb = k0 + ((lane >> 4) << 3);
    const _Float16* p = base + (size_t)r * ld + kb;
    v8h lo = *(const v8h*)(p);
    v8h hi = *(const v8h*)(p + 16);
    v16h f;
#pragma unroll
    for (int i = 0; i < 8; ++i) { f[i] = lo[i]; f[i + 8] = hi[i]; }
    return f;
}

#define WMMA_F16(a, b, c) \
    __builtin_amdgcn_wmma_f32_16x16x32_f16(false, (a), false, (b), (short)0, (c), false, false)

// ---------------------------------------------------------------------------
// Elementwise helpers
// ---------------------------------------------------------------------------
__global__ void cvt2d_f32_f16(const float* __restrict__ src, _Float16* __restrict__ dst,
                              int rows, int cols, int ldd) {
    long total = (long)rows * ldd;
    for (long i = (long)blockIdx.x * blockDim.x + threadIdx.x; i < total;
         i += (long)gridDim.x * blockDim.x) {
        long r = i / ldd;
        int  c = (int)(i - r * ldd);
        dst[i] = (c < cols) ? (_Float16)src[r * cols + c] : (_Float16)0.0f;
    }
}

__global__ void zero_f16(_Float16* __restrict__ dst, long n) {
    for (long i = (long)blockIdx.x * blockDim.x + threadIdx.x; i < n;
         i += (long)gridDim.x * blockDim.x)
        dst[i] = (_Float16)0.0f;
}

__global__ void silu_mul_f16(_Float16* __restrict__ gate, const _Float16* __restrict__ up, long n) {
    for (long i = (long)blockIdx.x * blockDim.x + threadIdx.x; i < n;
         i += (long)gridDim.x * blockDim.x) {
        float g = (float)gate[i];
        float u = (float)up[i];
        float s = g / (1.0f + __expf(-g));
        gate[i] = (_Float16)(s * u);
    }
}

// RMSNorm: one block per token row, f32 in -> f16 out
__global__ void rmsnorm_f16(const float* __restrict__ x, const float* __restrict__ w,
                            _Float16* __restrict__ y, int D) {
    long row = blockIdx.x;
    const float* xr = x + row * (long)D;
    float s = 0.0f;
    for (int i = threadIdx.x; i < D; i += blockDim.x) {
        float v = xr[i];
        s += v * v;
    }
#pragma unroll
    for (int m = 16; m > 0; m >>= 1) s += __shfl_xor(s, m, 32);
    __shared__ float red[9];
    int wave = threadIdx.x >> 5;
    if ((threadIdx.x & 31) == 0) red[wave] = s;
    __syncthreads();
    if (threadIdx.x == 0) {
        float t = 0.0f;
        int nw = blockDim.x >> 5;
        for (int i = 0; i < nw; ++i) t += red[i];
        red[8] = rsqrtf(t / (float)D + 1e-5f);
    }
    __syncthreads();
    float rr = red[8];
    for (int i = threadIdx.x; i < D; i += blockDim.x)
        y[row * (long)D + i] = (_Float16)(xr[i] * rr * w[i]);
}

// RoPE in place on [nrows = nheads*ntok, 64] f16; pairs (j, j+32), j<32.
__global__ void rope_f16(_Float16* __restrict__ x, const int* __restrict__ pos_ids,
                         long nrows, int ntok, int T) {
    long total = nrows * 32;
    for (long i = (long)blockIdx.x * blockDim.x + threadIdx.x; i < total;
         i += (long)gridDim.x * blockDim.x) {
        int  j   = (int)(i & 31);
        long row = i >> 5;
        long tok = row % ntok;
        int  t   = (int)(tok % T);
        float pos = (float)pos_ids[t];
        _Float16* p = x + row * 64;
        float a0 = pos * __powf(10000.0f, -(float)(2 * (j >> 1)) / 64.0f);
        float a1 = pos * __powf(10000.0f, -(float)(2 * ((j + 32) >> 1)) / 64.0f);
        float x0 = (float)p[j];
        float x1 = (float)p[j + 32];
        p[j]      = (_Float16)(x0 * __cosf(a0) - x1 * __sinf(a0));
        p[j + 32] = (_Float16)(x1 * __cosf(a1) + x0 * __sinf(a1));
    }
}

// ---------------------------------------------------------------------------
// Generic WMMA GEMM:  C[M,N] (f16) = A[M,K] (f16, row-major) * B[N,K]^T
// Requires M % 128 == 0, K % 32 == 0, N % 16 == 0.
// Block = 256 threads = 8 waves (4x2); block tile 128(M) x 128(N); each wave
// owns a 32x64 tile -> 2 A-frags + 4 B-frags -> 8 WMMAs per k-step.
// Out-of-range N tiles clamp the B row to 0 (the accumulated garbage is
// never stored) so the k-loop has no EXEC-mask manipulation.  Prefetch
// offsets are clamped to stay in-bounds (branchless).
// Batched over gridDim.z with strides sA/sB/sC.
// ---------------------------------------------------------------------------
__global__ void __launch_bounds__(256)
gemm_nt_f16(const _Float16* __restrict__ A, const _Float16* __restrict__ B,
            _Float16* __restrict__ C, int M, int N, int K,
            int lda, int ldb, int ldc, long sA, long sB, long sC) {
    A += (size_t)blockIdx.z * sA;
    B += (size_t)blockIdx.z * sB;
    C += (size_t)blockIdx.z * sC;
    int lane = threadIdx.x & 31;
    int wave = threadIdx.x >> 5;
    int wm = wave >> 1, wn = wave & 1;
    int m0 = blockIdx.x * 128 + wm * 32;
    int n0 = blockIdx.y * 128 + wn * 64;
    int bn[4];
#pragma unroll
    for (int j = 0; j < 4; ++j) {
        int nj = n0 + j * 16;
        bn[j] = (nj < N) ? nj : 0;   // clamp: loads in-bounds, tile unused
    }

    v8f acc[2][4];
#pragma unroll
    for (int i = 0; i < 2; ++i)
#pragma unroll
        for (int j = 0; j < 4; ++j) acc[i][j] = zero8f();

    for (int k = 0; k < K; k += 32) {
        int kpf = k + 32 <= K - 32 ? k + 32 : K - 32;  // clamped prefetch
        __builtin_prefetch(A + (size_t)(m0 + (lane & 15)) * lda + kpf, 0, 3);
        __builtin_prefetch(B + (size_t)(bn[0] + (lane & 15)) * ldb + kpf, 0, 3);
        v16h a0 = load_frag_rowmajor(A, lda, m0, k, lane);
        v16h a1 = load_frag_rowmajor(A, lda, m0 + 16, k, lane);
        v16h b0 = load_frag_rowmajor(B, ldb, bn[0], k, lane);
        v16h b1 = load_frag_rowmajor(B, ldb, bn[1], k, lane);
        v16h b2 = load_frag_rowmajor(B, ldb, bn[2], k, lane);
        v16h b3 = load_frag_rowmajor(B, ldb, bn[3], k, lane);
        acc[0][0] = WMMA_F16(a0, b0, acc[0][0]);
        acc[0][1] = WMMA_F16(a0, b1, acc[0][1]);
        acc[0][2] = WMMA_F16(a0, b2, acc[0][2]);
        acc[0][3] = WMMA_F16(a0, b3, acc[0][3]);
        acc[1][0] = WMMA_F16(a1, b0, acc[1][0]);
        acc[1][1] = WMMA_F16(a1, b1, acc[1][1]);
        acc[1][2] = WMMA_F16(a1, b2, acc[1][2]);
        acc[1][3] = WMMA_F16(a1, b3, acc[1][3]);
    }

    int cn = lane & 15, half = lane >> 4;
#pragma unroll
    for (int i = 0; i < 2; ++i)
#pragma unroll
        for (int j = 0; j < 4; ++j) {
            int nb = n0 + j * 16;
            if (nb >= N) continue;
            _Float16* cp = C + (size_t)(m0 + i * 16 + half * 8) * ldc + nb + cn;
#pragma unroll
            for (int r = 0; r < 8; ++r) cp[(size_t)r * ldc] = (_Float16)acc[i][j][r];
        }
}

// Same GEMM but f32 output with residual add (resid may alias Cout).
__global__ void __launch_bounds__(256)
gemm_nt_resid_f32(const _Float16* __restrict__ A, const _Float16* __restrict__ B,
                  const float* __restrict__ resid, float* __restrict__ Cout,
                  int M, int N, int K, int lda, int ldb, int ldc) {
    int lane = threadIdx.x & 31;
    int wave = threadIdx.x >> 5;
    int wm = wave >> 1, wn = wave & 1;
    int m0 = blockIdx.x * 128 + wm * 32;
    int n0 = blockIdx.y * 128 + wn * 64;
    int bn[4];
#pragma unroll
    for (int j = 0; j < 4; ++j) {
        int nj = n0 + j * 16;
        bn[j] = (nj < N) ? nj : 0;
    }

    v8f acc[2][4];
#pragma unroll
    for (int i = 0; i < 2; ++i)
#pragma unroll
        for (int j = 0; j < 4; ++j) acc[i][j] = zero8f();

    for (int k = 0; k < K; k += 32) {
        int kpf = k + 32 <= K - 32 ? k + 32 : K - 32;
        __builtin_prefetch(A + (size_t)(m0 + (lane & 15)) * lda + kpf, 0, 3);
        __builtin_prefetch(B + (size_t)(bn[0] + (lane & 15)) * ldb + kpf, 0, 3);
        v16h a0 = load_frag_rowmajor(A, lda, m0, k, lane);
        v16h a1 = load_frag_rowmajor(A, lda, m0 + 16, k, lane);
        v16h b0 = load_frag_rowmajor(B, ldb, bn[0], k, lane);
        v16h b1 = load_frag_rowmajor(B, ldb, bn[1], k, lane);
        v16h b2 = load_frag_rowmajor(B, ldb, bn[2], k, lane);
        v16h b3 = load_frag_rowmajor(B, ldb, bn[3], k, lane);
        acc[0][0] = WMMA_F16(a0, b0, acc[0][0]);
        acc[0][1] = WMMA_F16(a0, b1, acc[0][1]);
        acc[0][2] = WMMA_F16(a0, b2, acc[0][2]);
        acc[0][3] = WMMA_F16(a0, b3, acc[0][3]);
        acc[1][0] = WMMA_F16(a1, b0, acc[1][0]);
        acc[1][1] = WMMA_F16(a1, b1, acc[1][1]);
        acc[1][2] = WMMA_F16(a1, b2, acc[1][2]);
        acc[1][3] = WMMA_F16(a1, b3, acc[1][3]);
    }

    int cn = lane & 15, half = lane >> 4;
#pragma unroll
    for (int i = 0; i < 2; ++i)
#pragma unroll
        for (int j = 0; j < 4; ++j) {
            int nb = n0 + j * 16;
            if (nb >= N) continue;
            size_t off = (size_t)(m0 + i * 16 + half * 8) * ldc + nb + cn;
#pragma unroll
            for (int r = 0; r < 8; ++r)
                Cout[off + (size_t)r * ldc] = resid[off + (size_t)r * ldc] + acc[i][j][r];
        }
}

// ---------------------------------------------------------------------------
// Fused causal flash-attention with GQA.
// q: [H,  B*T, 64] f16 (RoPE applied)   k,v: [HK, B*T, 64] f16
// out: [B*T, H*64] f16
// Block = 128 threads (4 waves); each wave handles 16 query rows; grid =
// (T/64, H, B).  Per 32-key step: 4 WMMAs for S = Q*K^T, online softmax with
// shfl row reductions, P staged through LDS (C-layout -> A-layout transpose),
// V tile (32x64) staged into LDS via GLOBAL_LOAD_ASYNC_TO_LDS_B128, then
// 4 WMMAs for O += P*V.
// ---------------------------------------------------------------------------
__global__ void __launch_bounds__(128)
attn_flash_f16(const _Float16* __restrict__ qb, const _Float16* __restrict__ kb,
               const _Float16* __restrict__ vb, _Float16* __restrict__ ob,
               int T, int H, int HK, int ntok) {
    const int h = blockIdx.y;
    const int b = blockIdx.z;
    const int wave = threadIdx.x >> 5;
    const int lane = threadIdx.x & 31;
    const int qbase = blockIdx.x * 64 + wave * 16;
    const int hk = h / (H / HK);
    const int half = lane >> 4;
    const int cn = lane & 15;

    const _Float16* qrow  = qb + ((size_t)h * ntok + b * T + qbase) * 64;
    const _Float16* kbase = kb + ((size_t)hk * ntok + b * T) * 64;
    const _Float16* vbase = vb + ((size_t)hk * ntok + b * T) * 64;

    v16h qa0 = load_frag_rowmajor(qrow, 64, 0, 0, lane);
    v16h qa1 = load_frag_rowmajor(qrow, 64, 0, 32, lane);

    v8f o[4];
#pragma unroll
    for (int nb = 0; nb < 4; ++nb) o[nb] = zero8f();
    float mrow[8], lrow[8];
#pragma unroll
    for (int r = 0; r < 8; ++r) { mrow[r] = -1e30f; lrow[r] = 0.0f; }

    __shared__ _Float16 plds[4][16][40];    // P transpose staging (per wave)
    __shared__ _Float16 vlds[4][32][64];    // async-copied V tile (per wave)
    _Float16(*pl)[40] = plds[wave];
    _Float16(*vt)[64] = vlds[wave];
    // Low 32 bits of a generic pointer into LDS == LDS byte offset
    // (flat->LDS aperture truncation), usable as the async-copy VDST operand.
    unsigned vt_off = (unsigned)(uintptr_t)(&vt[0][0]) + (unsigned)lane * 128u;

    const int qend = qbase + 15;
    const float scale = 0.125f; // 1/sqrt(64)

    for (int s0 = 0; s0 <= qend; s0 += 32) {
        // ---- kick off async V tile copy: 32 rows x 64 halfs, one row/lane --
        {
            const _Float16* gsrc = vbase + (size_t)(s0 + lane) * 64;
            // WAR: previous iteration's DS reads of vlds must drain first;
            // async LDS writes are not ordered against DS ops.
            asm volatile("s_wait_dscnt 0x0" ::: "memory");
#pragma unroll
            for (int j = 0; j < 8; ++j) {
                asm volatile("global_load_async_to_lds_b128 %0, %1, off"
                             :: "v"(vt_off + j * 16u), "v"(gsrc + j * 8)
                             : "memory");
            }
        }

        // ---- S = Q * K^T  (two 16x16 tiles over 32 keys) ----
        v8f st0 = zero8f(), st1 = zero8f();
        {
            v16h k00 = load_frag_rowmajor(kbase, 64, s0, 0, lane);
            v16h k01 = load_frag_rowmajor(kbase, 64, s0, 32, lane);
            st0 = WMMA_F16(qa0, k00, st0);
            st0 = WMMA_F16(qa1, k01, st0);
            v16h k10 = load_frag_rowmajor(kbase, 64, s0 + 16, 0, lane);
            v16h k11 = load_frag_rowmajor(kbase, 64, s0 + 16, 32, lane);
            st1 = WMMA_F16(qa0, k10, st1);
            st1 = WMMA_F16(qa1, k11, st1);
        }
        // ---- causal mask + online softmax ----
        float p0[8], p1[8], mx[8];
#pragma unroll
        for (int r = 0; r < 8; ++r) {
            int rowg = qbase + r + 8 * half;
            float v0 = ((s0 + cn) <= rowg) ? st0[r] * scale : -1e30f;
            float v1 = ((s0 + 16 + cn) <= rowg) ? st1[r] * scale : -1e30f;
            p0[r] = v0; p1[r] = v1;
            mx[r] = fmaxf(v0, v1);
        }
#pragma unroll
        for (int msk = 1; msk < 16; msk <<= 1)
#pragma unroll
            for (int r = 0; r < 8; ++r) mx[r] = fmaxf(mx[r], __shfl_xor(mx[r], msk, 32));
        float alpha[8], rs[8];
#pragma unroll
        for (int r = 0; r < 8; ++r) {
            float mn = fmaxf(mrow[r], mx[r]);
            alpha[r] = __expf(mrow[r] - mn);
            mrow[r] = mn;
            p0[r] = __expf(p0[r] - mn);
            p1[r] = __expf(p1[r] - mn);
            rs[r] = p0[r] + p1[r];
        }
#pragma unroll
        for (int msk = 1; msk < 16; msk <<= 1)
#pragma unroll
            for (int r = 0; r < 8; ++r) rs[r] += __shfl_xor(rs[r], msk, 32);
#pragma unroll
        for (int r = 0; r < 8; ++r) lrow[r] = lrow[r] * alpha[r] + rs[r];
#pragma unroll
        for (int nb = 0; nb < 4; ++nb)
#pragma unroll
            for (int r = 0; r < 8; ++r) o[nb][r] *= alpha[r];

        // ---- transpose P through LDS: C-layout -> A-layout ----
#pragma unroll
        for (int r = 0; r < 8; ++r) {
            pl[r + 8 * half][cn]      = (_Float16)p0[r];
            pl[r + 8 * half][16 + cn] = (_Float16)p1[r];
        }
        asm volatile("s_wait_dscnt 0x0" ::: "memory");
        v16h pa;
        {
            int m = lane & 15;
            int kb2 = half * 8;
            v8h lo = *(const v8h*)&pl[m][kb2];
            v8h hi = *(const v8h*)&pl[m][kb2 + 16];
#pragma unroll
            for (int i = 0; i < 8; ++i) { pa[i] = lo[i]; pa[i + 8] = hi[i]; }
        }

        // ---- O += P * V  (V gathered from the async-filled LDS tile) ----
        asm volatile("s_wait_asynccnt 0x0" ::: "memory");
#pragma unroll
        for (int nb = 0; nb < 4; ++nb) {
            v16h vf;
            int n = nb * 16 + cn;
#pragma unroll
            for (int j = 0; j < 8; ++j) {
                int k1 = half * 8 + j;
                vf[j]     = vt[k1][n];
                vf[j + 8] = vt[k1 + 16][n];
            }
            o[nb] = WMMA_F16(pa, vf, o[nb]);
        }
    }

    // ---- finalize and store ----
    float inv_l[8];
#pragma unroll
    for (int r = 0; r < 8; ++r) inv_l[r] = 1.0f / lrow[r];
    _Float16* op = ob + (size_t)(b * T + qbase + 8 * half) * (H * 64) + h * 64 + cn;
#pragma unroll
    for (int nb = 0; nb < 4; ++nb)
#pragma unroll
        for (int r = 0; r < 8; ++r)
            op[(size_t)r * (H * 64) + nb * 16] = (_Float16)(o[nb][r] * inv_l[r]);
}

// ---------------------------------------------------------------------------
// Host launcher
// ---------------------------------------------------------------------------
extern "C" void kernel_launch(void* const* d_in, const int* in_sizes, int n_in,
                              void* d_out, int out_size, void* d_ws, size_t ws_size,
                              hipStream_t stream) {
    (void)in_sizes; (void)n_in; (void)out_size; (void)ws_size;
    const int Bc = 2, Tc = 1024, Dc = 2048, Hc = 32, HKc = 8;
    const int ROc = 1024, RFFc = 1024, INTERc = 8192;
    const int NT = Bc * Tc; // 2048 tokens

    const float* x_f32   = (const float*)d_in[0];
    const int*   pos_ids = (const int*)d_in[1];
    const float* ln1_w   = (const float*)d_in[2];
    const float* ln2_w   = (const float*)d_in[3];
    float* out_f32 = (float*)d_out;

    // ---- workspace bump allocator ----
    char* wsp = (char*)d_ws;
    auto alloc = [&](size_t elems) -> _Float16* {
        _Float16* p = (_Float16*)wsp;
        wsp += ((elems * sizeof(_Float16)) + 255) & ~(size_t)255;
        return p;
    };
    _Float16* h16    = alloc((size_t)NT * Dc);
    _Float16* qV16   = alloc((size_t)Hc * 48 * Dc);
    _Float16* kV16   = alloc((size_t)HKc * 48 * Dc);
    _Float16* vV16   = alloc((size_t)HKc * 48 * Dc);
    _Float16* qUs16  = alloc((size_t)Hc * 64 * 64);
    _Float16* kUs16  = alloc((size_t)HKc * 64 * 64);
    _Float16* vUs16  = alloc((size_t)HKc * 64 * 64);
    _Float16* oV16   = alloc((size_t)ROc * Dc);
    _Float16* oUs16  = alloc((size_t)Dc * ROc);
    _Float16* gV16   = alloc((size_t)RFFc * Dc);
    _Float16* uV16   = alloc((size_t)RFFc * Dc);
    _Float16* gUs16  = alloc((size_t)INTERc * RFFc);
    _Float16* uUs16  = alloc((size_t)INTERc * RFFc);
    _Float16* dV16   = alloc((size_t)RFFc * INTERc);
    _Float16* dUs16  = alloc((size_t)Dc * RFFc);
    _Float16* xr_q   = alloc((size_t)Hc * NT * 64);
    _Float16* xr_k   = alloc((size_t)HKc * NT * 64);
    _Float16* xr_v   = alloc((size_t)HKc * NT * 64);
    _Float16* q_buf  = alloc((size_t)Hc * NT * 64);
    _Float16* k_buf  = alloc((size_t)HKc * NT * 64);
    _Float16* v_buf  = alloc((size_t)HKc * NT * 64);
    _Float16* attn16 = alloc((size_t)NT * Dc);
    _Float16* ar16   = alloc((size_t)NT * ROc);
    _Float16* h2_16  = alloc((size_t)NT * Dc);
    _Float16* gr16   = alloc((size_t)NT * RFFc);
    _Float16* ur16   = alloc((size_t)NT * RFFc);
    _Float16* gate16 = alloc((size_t)NT * INTERc);
    _Float16* up16   = alloc((size_t)NT * INTERc);
    _Float16* ir16   = alloc((size_t)NT * RFFc);

    auto gsblocks = [](long total) -> int {
        long b = (total + 255) / 256;
        return (int)(b > 32768 ? 32768 : b);
    };
    auto cvt = [&](int in_idx, _Float16* dst, int rows, int cols, int ldd) {
        long total = (long)rows * ldd;
        cvt2d_f32_f16<<<gsblocks(total), 256, 0, stream>>>((const float*)d_in[in_idx], dst,
                                                           rows, cols, ldd);
    };

    // ---- weight conversions (f32 -> f16, rank-48 Us padded to 64) ----
    cvt(5,  qV16,  Hc * 48, Dc, Dc);
    cvt(7,  kV16,  HKc * 48, Dc, Dc);
    cvt(9,  vV16,  HKc * 48, Dc, Dc);
    cvt(4,  qUs16, Hc * 64, 48, 64);
    cvt(6,  kUs16, HKc * 64, 48, 64);
    cvt(8,  vUs16, HKc * 64, 48, 64);
    cvt(11, oV16,  ROc, Dc, Dc);
    cvt(10, oUs16, Dc, ROc, ROc);
    cvt(13, gV16,  RFFc, Dc, Dc);
    cvt(15, uV16,  RFFc, Dc, Dc);
    cvt(12, gUs16, INTERc, RFFc, RFFc);
    cvt(14, uUs16, INTERc, RFFc, RFFc);
    cvt(17, dV16,  RFFc, INTERc, INTERc);
    cvt(16, dUs16, Dc, RFFc, RFFc);

    // zero the K-padded rank buffers (cols 48..63 must be 0 for next GEMM)
    zero_f16<<<gsblocks((long)Hc * NT * 64), 256, 0, stream>>>(xr_q, (long)Hc * NT * 64);
    zero_f16<<<gsblocks((long)HKc * NT * 64), 256, 0, stream>>>(xr_k, (long)HKc * NT * 64);
    zero_f16<<<gsblocks((long)HKc * NT * 64), 256, 0, stream>>>(xr_v, (long)HKc * NT * 64);

    auto gemm = [&](const _Float16* A, const _Float16* B, _Float16* C, int M, int N, int K,
                    int lda, int ldb, int ldc, long sA, long sB, long sC, int nbatch) {
        dim3 g(M / 128, (N + 127) / 128, nbatch);
        gemm_nt_f16<<<g, dim3(256), 0, stream>>>(A, B, C, M, N, K, lda, ldb, ldc, sA, sB, sC);
    };

    // ---- attention block ----
    rmsnorm_f16<<<NT, 256, 0, stream>>>(x_f32, ln1_w, h16, Dc);

    // xr = h * V^T per head  (N=48 into K-padded [NT,64] buffers)
    gemm(h16, qV16, xr_q, NT, 48, Dc, Dc, Dc, 64, 0, (long)48 * Dc, (long)NT * 64, Hc);
    gemm(h16, kV16, xr_k, NT, 48, Dc, Dc, Dc, 64, 0, (long)48 * Dc, (long)NT * 64, HKc);
    gemm(h16, vV16, xr_v, NT, 48, Dc, Dc, Dc, 64, 0, (long)48 * Dc, (long)NT * 64, HKc);

    // q/k/v = xr * Us^T per head  -> [head][NT][64]
    gemm(xr_q, qUs16, q_buf, NT, 64, 64, 64, 64, 64, (long)NT * 64, 64 * 64, (long)NT * 64, Hc);
    gemm(xr_k, kUs16, k_buf, NT, 64, 64, 64, 64, 64, (long)NT * 64, 64 * 64, (long)NT * 64, HKc);
    gemm(xr_v, vUs16, v_buf, NT, 64, 64, 64, 64, 64, (long)NT * 64, 64 * 64, (long)NT * 64, HKc);

    // RoPE on q and k
    rope_f16<<<gsblocks((long)Hc * NT * 32), 256, 0, stream>>>(q_buf, pos_ids, (long)Hc * NT, NT, Tc);
    rope_f16<<<gsblocks((long)HKc * NT * 32), 256, 0, stream>>>(k_buf, pos_ids, (long)HKc * NT, NT, Tc);

    // fused causal GQA flash attention -> attn16 [NT, H*64]
    attn_flash_f16<<<dim3(Tc / 64, Hc, Bc), dim3(128), 0, stream>>>(q_buf, k_buf, v_buf,
                                                                    attn16, Tc, Hc, HKc, NT);

    // o-proj (low rank) + residual -> d_out (f32)
    gemm(attn16, oV16, ar16, NT, ROc, Dc, Dc, Dc, ROc, 0, 0, 0, 1);
    gemm_nt_resid_f32<<<dim3(NT / 128, Dc / 128, 1), dim3(256), 0, stream>>>(
        ar16, oUs16, x_f32, out_f32, NT, Dc, ROc, ROc, ROc, Dc);

    // ---- SwiGLU MLP block ----
    rmsnorm_f16<<<NT, 256, 0, stream>>>(out_f32, ln2_w, h2_16, Dc);

    gemm(h2_16, gV16, gr16, NT, RFFc, Dc, Dc, Dc, RFFc, 0, 0, 0, 1);
    gemm(h2_16, uV16, ur16, NT, RFFc, Dc, Dc, Dc, RFFc, 0, 0, 0, 1);
    gemm(gr16, gUs16, gate16, NT, INTERc, RFFc, RFFc, RFFc, INTERc, 0, 0, 0, 1);
    gemm(ur16, uUs16, up16,   NT, INTERc, RFFc, RFFc, RFFc, INTERc, 0, 0, 0, 1);

    silu_mul_f16<<<gsblocks((long)NT * INTERc), 256, 0, stream>>>(gate16, up16, (long)NT * INTERc);

    gemm(gate16, dV16, ir16, NT, RFFc, INTERc, INTERc, INTERc, RFFc, 0, 0, 0, 1);
    gemm_nt_resid_f32<<<dim3(NT / 128, Dc / 128, 1), dim3(256), 0, stream>>>(
        ir16, dUs16, out_f32, out_f32, NT, Dc, RFFc, RFFc, RFFc, Dc);
}
